// GCN_71725953843990
// MI455X (gfx1250) — matrix operand order repaired and checked by reading, not compiled, not run on hardware
//
#include <hip/hip_runtime.h>
#include <hip/hip_bf16.h>

typedef __attribute__((ext_vector_type(16))) __bf16 v16bf;
typedef __attribute__((ext_vector_type(8)))  __bf16 v8bf;
typedef __attribute__((ext_vector_type(8)))  float  v8f;

#define N_NODES 8192
#define DIM     512

// workspace layout (bytes)
#define OFF_DINV 0                                   // 8192 * 4        = 32 KB
#define OFF_WT1  (32768)                             // 512*512*2       = 512 KB
#define OFF_WT2  (32768 + 524288)                    // 512*512*2       = 512 KB
#define OFF_SPT  (2u*1024u*1024u)                    // 512*8192*2      = 8 MB (reused)
#define OFF_H1   (OFF_SPT + (size_t)DIM*N_NODES*2)   // 8192*512*2      = 8 MB

// ---------------------------------------------------------------------------
// d_inv[i] = rsqrt(sum_j A[i][j])   (one block per row)
// ---------------------------------------------------------------------------
__global__ void __launch_bounds__(256) rowsum_rsqrt(const float* __restrict__ A,
                                                    float* __restrict__ dinv) {
  const int row = blockIdx.x;
  const float4* ap = (const float4*)(A + (size_t)row * N_NODES);
  float s = 0.f;
  for (int i = threadIdx.x; i < N_NODES / 4; i += 256) {
    float4 v = ap[i];
    s += v.x + v.y + v.z + v.w;
  }
  __shared__ float red[256];
  red[threadIdx.x] = s;
  __syncthreads();
  for (int off = 128; off > 0; off >>= 1) {
    if (threadIdx.x < off) red[threadIdx.x] += red[threadIdx.x + off];
    __syncthreads();
  }
  if (threadIdx.x == 0) dinv[row] = rsqrtf(red[0]);
}

// ---------------------------------------------------------------------------
// WT[n][k] = (bf16) W[k][n]   (16x16 tiles through LDS)
// ---------------------------------------------------------------------------
__global__ void __launch_bounds__(256) transpose_cvt(const float* __restrict__ W,
                                                     __bf16* __restrict__ WT) {
  __shared__ float tile[16][17];
  const int k = blockIdx.y * 16 + threadIdx.y;
  const int n = blockIdx.x * 16 + threadIdx.x;
  tile[threadIdx.y][threadIdx.x] = W[(size_t)k * DIM + n];
  __syncthreads();
  const int nn = blockIdx.x * 16 + threadIdx.y;
  const int kk = blockIdx.y * 16 + threadIdx.x;
  WT[(size_t)nn * DIM + kk] = (__bf16)tile[threadIdx.x][threadIdx.y];
}

// ---------------------------------------------------------------------------
// WMMA helper
// ---------------------------------------------------------------------------
__device__ __forceinline__ v8f wmma_bf16(v16bf a, v16bf b, v8f c) {
  return __builtin_amdgcn_wmma_f32_16x16x32_bf16(
      /*neg_a=*/false, a, /*neg_b=*/false, b,
      /*c_mod=*/(short)0, c, /*reuse_a=*/false, /*reuse_b=*/false);
}

// ---------------------------------------------------------------------------
// Pipelined fragment staging.
// A-fragment (16x32 bf16): lane -> K {klo..klo+15}, klo = 16*(lane>=16)
// ---------------------------------------------------------------------------
struct ARaw { float4 r0, r1, r2, r3; };      // 16 fp32 (pre-conversion)

__device__ __forceinline__ ARaw load_araw_f32(const float* __restrict__ p, int k0) {
  // p = row base + klo
  ARaw a;
  const float4* p0 = (const float4*)(p + k0);
  const float4* p1 = (const float4*)(p + k0 + 8);
  a.r0 = p0[0]; a.r1 = p0[1]; a.r2 = p1[0]; a.r3 = p1[1];
  return a;
}

__device__ __forceinline__ v16bf cvt_araw(ARaw x) {
  v16bf a;
  a[0]  = (__bf16)x.r0.x; a[1]  = (__bf16)x.r0.y; a[2]  = (__bf16)x.r0.z; a[3]  = (__bf16)x.r0.w;
  a[4]  = (__bf16)x.r1.x; a[5]  = (__bf16)x.r1.y; a[6]  = (__bf16)x.r1.z; a[7]  = (__bf16)x.r1.w;
  a[8]  = (__bf16)x.r2.x; a[9]  = (__bf16)x.r2.y; a[10] = (__bf16)x.r2.z; a[11] = (__bf16)x.r2.w;
  a[12] = (__bf16)x.r3.x; a[13] = (__bf16)x.r3.y; a[14] = (__bf16)x.r3.z; a[15] = (__bf16)x.r3.w;
  return a;
}

struct ABf { v8bf lo, hi; };                 // 16 bf16 A-fragment halves

__device__ __forceinline__ ABf load_araw_bf16(const __bf16* __restrict__ p, int k0) {
  ABf a;
  a.lo = *(const v8bf*)(p + k0);
  a.hi = *(const v8bf*)(p + k0 + 8);
  return a;
}

__device__ __forceinline__ v16bf join_abf(ABf x) {
  v16bf a;
#pragma unroll
  for (int j = 0; j < 8; ++j) { a[j] = x.lo[j]; a[8 + j] = x.hi[j]; }
  return a;
}

struct BFr { v16bf b0, b1, b2, b3; };        // 4 B-fragments (32x16 bf16 each)

__device__ __forceinline__ BFr load_bfr(const __bf16* __restrict__ p0,
                                        const __bf16* __restrict__ p1,
                                        const __bf16* __restrict__ p2,
                                        const __bf16* __restrict__ p3, int k0) {
  BFr b;
  b.b0 = *(const v16bf*)(p0 + k0);
  b.b1 = *(const v16bf*)(p1 + k0);
  b.b2 = *(const v16bf*)(p2 + k0);
  b.b3 = *(const v16bf*)(p3 + k0);
  return b;
}

// ---------------------------------------------------------------------------
// Big GEMM:  out[m][n] = d[m] * sum_k (bf16)A[m][k] * SpT[n][k]  + bias[n]
// A: fp32 [8192][8192] (bf16 on the fly); SpT: bf16 [512][8192] (k-contiguous)
// LAYER1: relu + bf16 store to Hout;  else fp32 store to Fout.
// Block = 8 waves -> 128 rows x 64 cols; wave tile 16x64.
// Software-pipelined: iteration i issues loads for i+1, computes on i.
// ---------------------------------------------------------------------------
template <bool LAYER1>
__global__ void __launch_bounds__(256)
gemm_big(const float* __restrict__ A, const __bf16* __restrict__ SpT,
         const float* __restrict__ dinv, const float* __restrict__ bias,
         __bf16* __restrict__ Hout, float* __restrict__ Fout) {
  const int lane = threadIdx.x & 31;
  const int wave = threadIdx.x >> 5;
  const int half = lane >> 4;
  const int l    = lane & 15;
  const int m0   = blockIdx.x * 128 + wave * 16;
  const int n0   = blockIdx.y * 64;
  const int row  = m0 + l;
  const int klo  = half * 16;

  const float*  abase = A + (size_t)row * N_NODES + klo;
  const __bf16* bp0   = SpT + (size_t)(n0 +  0 + l) * N_NODES + half * 16;
  const __bf16* bp1   = SpT + (size_t)(n0 + 16 + l) * N_NODES + half * 16;
  const __bf16* bp2   = SpT + (size_t)(n0 + 32 + l) * N_NODES + half * 16;
  const __bf16* bp3   = SpT + (size_t)(n0 + 48 + l) * N_NODES + half * 16;

  v8f c0 = {}, c1 = {}, c2 = {}, c3 = {};

  // pipeline prologue
  ARaw ar = load_araw_f32(abase, 0);
  BFr  br = load_bfr(bp0, bp1, bp2, bp3, 0);

  for (int k0 = 32; k0 < N_NODES; k0 += 32) {
    __builtin_prefetch(abase + k0 + 2048, 0, 3);  // stream A ~8KB ahead
    ARaw arn = load_araw_f32(abase, k0);          // loads for next step
    BFr  brn = load_bfr(bp0, bp1, bp2, bp3, k0);
    v16bf a = cvt_araw(ar);                       // compute on current step
    c0 = wmma_bf16(a, br.b0, c0);
    c1 = wmma_bf16(a, br.b1, c1);
    c2 = wmma_bf16(a, br.b2, c2);
    c3 = wmma_bf16(a, br.b3, c3);
    ar = arn; br = brn;
  }
  {  // pipeline epilogue
    v16bf a = cvt_araw(ar);
    c0 = wmma_bf16(a, br.b0, c0);
    c1 = wmma_bf16(a, br.b1, c1);
    c2 = wmma_bf16(a, br.b2, c2);
    c3 = wmma_bf16(a, br.b3, c3);
  }

  // C layout: VGPR r -> (M = r + 8*half, N = l)
  const int mbase = m0 + 8 * half;
#pragma unroll
  for (int r = 0; r < 8; ++r) {
    const float dv = dinv[mbase + r];
    const float vals[4] = {c0[r], c1[r], c2[r], c3[r]};
#pragma unroll
    for (int t = 0; t < 4; ++t) {
      const int col = n0 + t * 16 + l;
      float v = dv * vals[t] + bias[col];
      if constexpr (LAYER1) {
        v = v > 0.f ? v : 0.f;
        Hout[(size_t)(mbase + r) * DIM + col] = (__bf16)v;
      } else {
        Fout[(size_t)(mbase + r) * DIM + col] = v;
      }
    }
  }
}

// ---------------------------------------------------------------------------
// Small GEMM:  SpT[n][m] = (bf16)( d[m] * sum_k Aop[m][k] * WT[n][k] )
// A_F32: Aop = X fp32 [8192][512];  else Aop = H1 bf16 [8192][512].
// WT: bf16 [512][512] (k-contiguous per n).  Output transposed + d-scaled.
// ---------------------------------------------------------------------------
template <bool A_F32>
__global__ void __launch_bounds__(256)
gemm_small(const float* __restrict__ Af, const __bf16* __restrict__ Ab,
           const __bf16* __restrict__ WT, const float* __restrict__ dinv,
           __bf16* __restrict__ SpT) {
  const int lane = threadIdx.x & 31;
  const int wave = threadIdx.x >> 5;
  const int half = lane >> 4;
  const int l    = lane & 15;
  const int m0   = blockIdx.x * 128 + wave * 16;
  const int n0   = blockIdx.y * 64;
  const int row  = m0 + l;
  const int klo  = half * 16;

  const float*  afb = A_F32 ? (Af + (size_t)row * DIM + klo) : nullptr;
  const __bf16* abb = A_F32 ? nullptr : (Ab + (size_t)row * DIM + klo);
  const __bf16* bp0 = WT + (size_t)(n0 +  0 + l) * DIM + half * 16;
  const __bf16* bp1 = WT + (size_t)(n0 + 16 + l) * DIM + half * 16;
  const __bf16* bp2 = WT + (size_t)(n0 + 32 + l) * DIM + half * 16;
  const __bf16* bp3 = WT + (size_t)(n0 + 48 + l) * DIM + half * 16;

  v8f c0 = {}, c1 = {}, c2 = {}, c3 = {};

  ARaw arf; ABf arb;
  if constexpr (A_F32) arf = load_araw_f32(afb, 0); else arb = load_araw_bf16(abb, 0);
  BFr br = load_bfr(bp0, bp1, bp2, bp3, 0);

  for (int k0 = 32; k0 < DIM; k0 += 32) {
    ARaw arfn; ABf arbn;
    if constexpr (A_F32) arfn = load_araw_f32(afb, k0); else arbn = load_araw_bf16(abb, k0);
    BFr brn = load_bfr(bp0, bp1, bp2, bp3, k0);
    v16bf a = A_F32 ? cvt_araw(arf) : join_abf(arb);
    c0 = wmma_bf16(a, br.b0, c0);
    c1 = wmma_bf16(a, br.b1, c1);
    c2 = wmma_bf16(a, br.b2, c2);
    c3 = wmma_bf16(a, br.b3, c3);
    arf = arfn; arb = arbn; br = brn;
  }
  {
    v16bf a = A_F32 ? cvt_araw(arf) : join_abf(arb);
    c0 = wmma_bf16(a, br.b0, c0);
    c1 = wmma_bf16(a, br.b1, c1);
    c2 = wmma_bf16(a, br.b2, c2);
    c3 = wmma_bf16(a, br.b3, c3);
  }

  // transposed, d-scaled bf16 store: SpT[n][m]
  const int mbase = m0 + 8 * half;
#pragma unroll
  for (int t = 0; t < 4; ++t) {
    const int n = n0 + t * 16 + l;
    __bf16* op = SpT + (size_t)n * N_NODES + mbase;
    const v8f c = (t == 0) ? c0 : (t == 1) ? c1 : (t == 2) ? c2 : c3;
#pragma unroll
    for (int r = 0; r < 8; ++r)
      op[r] = (__bf16)(dinv[mbase + r] * c[r]);
  }
}

// ---------------------------------------------------------------------------
extern "C" void kernel_launch(void* const* d_in, const int* in_sizes, int n_in,
                              void* d_out, int out_size, void* d_ws, size_t ws_size,
                              hipStream_t stream) {
  (void)in_sizes; (void)n_in; (void)out_size; (void)ws_size;
  const float* X  = (const float*)d_in[0];  // [8192, 512]
  const float* A  = (const float*)d_in[1];  // [8192, 8192]
  const float* W1 = (const float*)d_in[2];  // [512, 512]
  const float* b1 = (const float*)d_in[3];  // [512]
  const float* W2 = (const float*)d_in[4];  // [512, 512]
  const float* b2 = (const float*)d_in[5];  // [512]
  float* out = (float*)d_out;               // [8192, 512]

  char*   ws   = (char*)d_ws;
  float*  dinv = (float*)(ws + OFF_DINV);
  __bf16* WT1  = (__bf16*)(ws + OFF_WT1);
  __bf16* WT2  = (__bf16*)(ws + OFF_WT2);
  __bf16* SpT  = (__bf16*)(ws + OFF_SPT);
  __bf16* H1   = (__bf16*)(ws + OFF_H1);

  rowsum_rsqrt<<<N_NODES, 256, 0, stream>>>(A, dinv);

  dim3 tb(16, 16), tg(DIM / 16, DIM / 16);
  transpose_cvt<<<tg, tb, 0, stream>>>(W1, WT1);
  transpose_cvt<<<tg, tb, 0, stream>>>(W2, WT2);

  dim3 gg(N_NODES / 128, DIM / 64);
  // layer 1: SpT = (d .* (X@W1))^T ; H1 = relu(d .* (A @ SpT^T) + b1)
  gemm_small<true ><<<gg, 256, 0, stream>>>(X, nullptr, WT1, dinv, SpT);
  gemm_big  <true ><<<gg, 256, 0, stream>>>(A, SpT, dinv, b1, H1, nullptr);
  // layer 2: SpT = (d .* (H1@W2))^T ; out = d .* (A @ SpT^T) + b2
  gemm_small<false><<<gg, 256, 0, stream>>>(nullptr, H1, WT2, dinv, SpT);
  gemm_big  <false><<<gg, 256, 0, stream>>>(A, SpT, dinv, b2, nullptr, out);
}
